// MultiHeadSelfAttention_61031485276804
// MI455X (gfx1250) — matrix operand order repaired and checked by reading, not compiled, not run on hardware
//
#include <hip/hip_runtime.h>
#include <hip/hip_bf16.h>
#include <cstdint>

#define DMODEL 1024
#define NHEADS 16
#define DHEAD  64
#define SEQ    2048
#define BATCH  4
#define MROWS  (BATCH * SEQ)   // 8192

typedef __attribute__((ext_vector_type(16))) __bf16 v16bf;
typedef __attribute__((ext_vector_type(8)))  float  v8f;

union FragU {
  v16bf v;
  uint4 q[2];
};

__device__ __forceinline__ unsigned short f2bf(float x) {
  unsigned int u = __float_as_uint(x);
  u += 0x7FFFu + ((u >> 16) & 1u);          // round-to-nearest-even
  return (unsigned short)(u >> 16);
}

// A fragment: 16(M) x 32(K) bf16. lane = M (mod 16), half h = lane>>4.
// Per ISA 7.12.2: lane holds K in {h*8+0..7} U {16+h*8+0..7}  -> two b128 loads.
__device__ __forceinline__ v16bf load_frag_a(const unsigned short* base, int ld) {
  const int lane = threadIdx.x & 31;
  const int m = lane & 15, h = lane >> 4;
  const unsigned short* p = base + m * ld + h * 8;
  FragU f;
  f.q[0] = *(const uint4*)(p);
  f.q[1] = *(const uint4*)(p + 16);
  return f.v;
}

// B fragment: 32(K) x 16(N) bf16. lane = N (mod 16); lane holds K in {h*8+0..7} U {16+h*8+0..7}.
// `base` points at (k=0, n=0) of a buffer storing B^T row-major (N rows of K
// contiguous values), stride ld.
__device__ __forceinline__ v16bf load_frag_b(const unsigned short* base, int ld) {
  const int lane = threadIdx.x & 31;
  const int n = lane & 15, h = lane >> 4;
  const unsigned short* p = base + n * ld + h * 8;
  FragU f;
  f.q[0] = *(const uint4*)(p);
  f.q[1] = *(const uint4*)(p + 16);
  return f.v;
}

__device__ __forceinline__ v8f wmma_bf16(v16bf a, v16bf b, v8f c) {
  return __builtin_amdgcn_wmma_f32_16x16x32_bf16(false, a, false, b, (short)0, c,
                                                 false, false);
}

__device__ __forceinline__ float rowmax16(float x) {
  x = fmaxf(x, __shfl_xor(x, 1, 32));
  x = fmaxf(x, __shfl_xor(x, 2, 32));
  x = fmaxf(x, __shfl_xor(x, 4, 32));
  x = fmaxf(x, __shfl_xor(x, 8, 32));
  return x;
}
__device__ __forceinline__ float rowsum16(float x) {
  x += __shfl_xor(x, 1, 32);
  x += __shfl_xor(x, 2, 32);
  x += __shfl_xor(x, 4, 32);
  x += __shfl_xor(x, 8, 32);
  return x;
}

// Async copy: one 16B chunk per lane, global -> LDS, tracked by ASYNCcnt.
__device__ __forceinline__ void async_b128_to_lds(unsigned lds_off,
                                                  const unsigned short* gaddr) {
  asm volatile("global_load_async_to_lds_b128 %0, %1, off"
               :: "v"(lds_off), "v"(gaddr)
               : "memory");
}
__device__ __forceinline__ void wait_asynccnt0() {
  asm volatile("s_wait_asynccnt 0x0" ::: "memory");
}

// ---------------------------------------------------------------------------
// fp32 -> bf16 conversion, 4 elements per thread
// ---------------------------------------------------------------------------
__global__ __launch_bounds__(256) void convert_f32_bf16(
    const float* __restrict__ in, unsigned short* __restrict__ out, int n4) {
  int i = blockIdx.x * blockDim.x + threadIdx.x;
  if (i >= n4) return;
  float4 f = ((const float4*)in)[i];
  unsigned long long pk = (unsigned long long)f2bf(f.x)
                        | ((unsigned long long)f2bf(f.y) << 16)
                        | ((unsigned long long)f2bf(f.z) << 32)
                        | ((unsigned long long)f2bf(f.w) << 48);
  ((unsigned long long*)out)[i] = pk;
}

// ---------------------------------------------------------------------------
// Y = X @ W^T.  X:[8192,1024] bf16 row-major, W:[1024,1024] bf16 row-major.
// Block (8 waves) owns a 256(M) x 64(N) macro-tile. The 64x32 W tile for each
// K-step is staged into LDS with GLOBAL_LOAD_ASYNC_TO_LDS_B128, double
// buffered (copy of tile k+1 overlaps WMMA on tile k). Each wave computes a
// 32(M) x 64(N) tile: 2 A-frags, 8 C-frags, 8 WMMAs per K-step.
// mode 0/1: store [B,H,T,Dh] bf16 (Q,K)   mode 2: store [B,H,Dh,T] bf16 (V^T)
// mode 3: store fp32 row-major [M,D]
// ---------------------------------------------------------------------------
__global__ __launch_bounds__(256) void proj_gemm(
    const unsigned short* __restrict__ X, const unsigned short* __restrict__ W,
    unsigned short* __restrict__ outb, float* __restrict__ outf, int mode) {
  const int K = DMODEL, N = DMODEL;
  __shared__ __align__(16) unsigned short smem[2][64][32];   // 2 x 4KB W tiles

  const int bid = blockIdx.x;                 // 512 blocks
  const int n0 = (bid & 15) << 6;             // 16 N groups of 64
  const int m0 = ((bid >> 4) << 8) + ((threadIdx.x >> 5) << 5);  // wave's 32 rows

  // staging assignment: thread t copies row (t>>2), 16B chunk (t&3)
  const int srow = threadIdx.x >> 2, schunk = threadIdx.x & 3;
  const unsigned short* gW = W + (size_t)(n0 + srow) * K + schunk * 8;
  unsigned lds0 = (unsigned)(size_t)(&smem[0][srow][schunk * 8]);
  unsigned lds1 = (unsigned)(size_t)(&smem[1][srow][schunk * 8]);

  const unsigned short* Xrow = X + (size_t)m0 * K;

  v8f c[2][4] = {};

  async_b128_to_lds(lds0, gW);                // stage tile 0
  wait_asynccnt0();
  __syncthreads();

  const int KT = K / 32;                      // 32 K-steps
  for (int kt = 0; kt < KT; ++kt) {
    const int cur = kt & 1;
    if (kt + 1 < KT)                          // stage tile kt+1 into other buf
      async_b128_to_lds(cur ? lds0 : lds1, gW + (kt + 1) * 32);

    const int k0 = kt * 32;
    __builtin_prefetch(Xrow + k0 + 128, 0, 1);
    v16bf a0 = load_frag_a(Xrow + k0, K);
    v16bf a1 = load_frag_a(Xrow + 16 * K + k0, K);
    const unsigned short* ldsW = &smem[cur][0][0];
#pragma unroll
    for (int j = 0; j < 4; ++j) {
      v16bf b = load_frag_b(ldsW + (16 * j) * 32, 32);
      c[0][j] = wmma_bf16(a0, b, c[0][j]);
      c[1][j] = wmma_bf16(a1, b, c[1][j]);
    }

    if (kt + 1 < KT) {
      wait_asynccnt0();                       // tile kt+1 landed
      __syncthreads();                        // everyone done reading buf cur
    }
  }

  const int lane = threadIdx.x & 31;
  const int ncol = lane & 15, h = lane >> 4;
#pragma unroll
  for (int part = 0; part < 2; ++part) {
#pragma unroll
    for (int j = 0; j < 4; ++j) {
#pragma unroll
      for (int v = 0; v < 8; ++v) {
        int m = m0 + part * 16 + h * 8 + v;   // global row (b*T + t)
        int feat = n0 + j * 16 + ncol;        // output feature
        float val = c[part][j][v];
        if (mode == 3) {
          outf[(size_t)m * N + feat] = val;
        } else {
          int b_ = m >> 11, t = m & (SEQ - 1);
          int head = feat >> 6, d = feat & 63;
          if (mode == 2)   // V transposed: [B,H,Dh,T]
            outb[(((size_t)(b_ * NHEADS + head) * DHEAD + d) * SEQ) + t] = f2bf(val);
          else             // Q/K: [B,H,T,Dh]
            outb[(((size_t)(b_ * NHEADS + head) * SEQ + t) * DHEAD) + d] = f2bf(val);
        }
      }
    }
  }
}

// ---------------------------------------------------------------------------
// Flash attention. One wave = one (b,h) x 16 query rows. Online softmax,
// P re-shaped C-layout -> A-layout via a 1KB per-wave LDS tile.
// Q,K: [B,H,T,64] bf16, Vt: [B,H,64,T] bf16, merged: [B,T,1024] bf16
// ---------------------------------------------------------------------------
__global__ __launch_bounds__(256) void attn_kernel(
    const unsigned short* __restrict__ Q, const unsigned short* __restrict__ Km,
    const unsigned short* __restrict__ Vt, unsigned short* __restrict__ merged) {
  __shared__ __align__(16) unsigned short plds[8][16][32];

  int wid = (blockIdx.x * blockDim.x + threadIdx.x) >> 5;  // 8192 waves
  int wIn = threadIdx.x >> 5;
  int q0 = (wid & 127) << 4;          // 128 query tiles of 16
  int bh = wid >> 7;                  // 64 (b,h) pairs
  const unsigned short* Qb = Q + ((size_t)bh * SEQ + q0) * DHEAD;
  const unsigned short* Kb = Km + (size_t)bh * SEQ * DHEAD;
  const unsigned short* Vb = Vt + (size_t)bh * DHEAD * SEQ;

  const int lane = threadIdx.x & 31;
  const int ncol = lane & 15, h = lane >> 4;

  // Q A-fragments, reused across all key blocks
  v16bf aq0 = load_frag_a(Qb, DHEAD);        // d = 0..31
  v16bf aq1 = load_frag_a(Qb + 32, DHEAD);   // d = 32..63

  v8f o[4] = {};
  float mrow[8], lrow[8];
#pragma unroll
  for (int v = 0; v < 8; ++v) { mrow[v] = -1e30f; lrow[v] = 0.0f; }

  for (int kb = 0; kb < q0 + 16; kb += 32) {
    v8f z0 = {}, z1 = {};
    v8f s0 = wmma_bf16(aq0, load_frag_b(Kb + (size_t)kb * DHEAD, DHEAD), z0);
    s0 = wmma_bf16(aq1, load_frag_b(Kb + (size_t)kb * DHEAD + 32, DHEAD), s0);
    v8f s1 = wmma_bf16(aq0, load_frag_b(Kb + (size_t)(kb + 16) * DHEAD, DHEAD), z1);
    s1 = wmma_bf16(aq1, load_frag_b(Kb + (size_t)(kb + 16) * DHEAD + 32, DHEAD), s1);

    int key0 = kb + ncol, key1 = kb + 16 + ncol;
#pragma unroll
    for (int v = 0; v < 8; ++v) {
      int qrow = q0 + h * 8 + v;
      float e0 = s0[v] * 0.125f;              // 1/sqrt(64)
      float e1 = s1[v] * 0.125f;
      if (key0 > qrow) e0 = -1e30f;           // causal mask
      if (key1 > qrow) e1 = -1e30f;
      float mblk = rowmax16(fmaxf(e0, e1));
      float mnew = fmaxf(mrow[v], mblk);
      float sc = __expf(mrow[v] - mnew);
      float p0 = __expf(e0 - mnew);
      float p1 = __expf(e1 - mnew);
      lrow[v] = lrow[v] * sc + rowsum16(p0 + p1);
      mrow[v] = mnew;
      o[0][v] *= sc; o[1][v] *= sc; o[2][v] *= sc; o[3][v] *= sc;
      plds[wIn][h * 8 + v][ncol] = f2bf(p0);
      plds[wIn][h * 8 + v][16 + ncol] = f2bf(p1);
    }
    __builtin_amdgcn_wave_barrier();   // keep DS stores before DS loads (in-order per wave)

    v16bf pA = load_frag_a(&plds[wIn][0][0], 32);
#pragma unroll
    for (int j = 0; j < 4; ++j) {      // N = d chunks of 16 over Dh=64
      v16bf bv = load_frag_b(Vb + (size_t)(j * 16) * SEQ + kb, SEQ);
      o[j] = wmma_bf16(pA, bv, o[j]);
    }
    __builtin_amdgcn_wave_barrier();
  }

  int b_ = bh >> 4, head = bh & 15;
#pragma unroll
  for (int v = 0; v < 8; ++v) {
    float inv = 1.0f / lrow[v];
    int t = q0 + h * 8 + v;
    unsigned short* dst =
        merged + ((size_t)(b_ * SEQ + t)) * DMODEL + head * DHEAD + ncol;
    dst[0]  = f2bf(o[0][v] * inv);
    dst[16] = f2bf(o[1][v] * inv);
    dst[32] = f2bf(o[2][v] * inv);
    dst[48] = f2bf(o[3][v] * inv);
  }
}

// ---------------------------------------------------------------------------
extern "C" void kernel_launch(void* const* d_in, const int* in_sizes, int n_in,
                              void* d_out, int out_size, void* d_ws, size_t ws_size,
                              hipStream_t stream) {
  const float* x  = (const float*)d_in[0];
  const float* Wq = (const float*)d_in[1];
  const float* Wk = (const float*)d_in[2];
  const float* Wv = (const float*)d_in[3];
  const float* Wo = (const float*)d_in[4];
  float* out = (float*)d_out;

  char* wsp = (char*)d_ws;
  size_t off = 0;
  auto take = [&](size_t bytes) -> unsigned short* {
    unsigned short* p = (unsigned short*)(wsp + off);
    off = (off + bytes + 255) & ~(size_t)255;
    return p;
  };
  unsigned short* xb     = take((size_t)MROWS * DMODEL * 2);   // 16 MB
  unsigned short* wqb    = take((size_t)DMODEL * DMODEL * 2);  //  2 MB
  unsigned short* wkb    = take((size_t)DMODEL * DMODEL * 2);
  unsigned short* wvb    = take((size_t)DMODEL * DMODEL * 2);
  unsigned short* wob    = take((size_t)DMODEL * DMODEL * 2);
  unsigned short* qbuf   = take((size_t)MROWS * DMODEL * 2);   // [B,H,T,Dh]
  unsigned short* kbuf   = take((size_t)MROWS * DMODEL * 2);   // [B,H,T,Dh]
  unsigned short* vbuf   = take((size_t)MROWS * DMODEL * 2);   // [B,H,Dh,T]
  unsigned short* merged = take((size_t)MROWS * DMODEL * 2);   // [B,T,D]

  // fp32 -> bf16
  {
    int n4 = MROWS * DMODEL / 4;
    convert_f32_bf16<<<(n4 + 255) / 256, 256, 0, stream>>>(x, xb, n4);
    int w4 = DMODEL * DMODEL / 4;
    convert_f32_bf16<<<(w4 + 255) / 256, 256, 0, stream>>>(Wq, wqb, w4);
    convert_f32_bf16<<<(w4 + 255) / 256, 256, 0, stream>>>(Wk, wkb, w4);
    convert_f32_bf16<<<(w4 + 255) / 256, 256, 0, stream>>>(Wv, wvb, w4);
    convert_f32_bf16<<<(w4 + 255) / 256, 256, 0, stream>>>(Wo, wob, w4);
  }

  // Q/K/V projections (512 blocks x 8 waves, 32x64 per wave)
  proj_gemm<<<512, 256, 0, stream>>>(xb, wqb, qbuf, nullptr, 0);
  proj_gemm<<<512, 256, 0, stream>>>(xb, wkb, kbuf, nullptr, 1);
  proj_gemm<<<512, 256, 0, stream>>>(xb, wvb, vbuf, nullptr, 2);

  // causal flash attention (8192 waves)
  attn_kernel<<<1024, 256, 0, stream>>>(qbuf, kbuf, vbuf, merged);

  // output projection -> fp32 d_out
  proj_gemm<<<512, 256, 0, stream>>>(merged, wob, nullptr, out, 3);

  (void)in_sizes; (void)n_in; (void)out_size; (void)ws_size;
}
